// MultiheadAttention_10711648436710
// MI455X (gfx1250) — matrix operand order repaired and checked by reading, not compiled, not run on hardware
//
#include <hip/hip_runtime.h>

typedef __attribute__((ext_vector_type(16))) _Float16 v16h;
typedef __attribute__((ext_vector_type(8)))  _Float16 v8h;
typedef __attribute__((ext_vector_type(8)))  float    v8f;
typedef __attribute__((ext_vector_type(4)))  float    v4f;

// ---------------------------------------------------------------------------
// CDNA5 WMMA f16 fragment layouts (CDNA5 ISA 7.12.2):
//  A (16x32): lane -> row M = lane&15, halfsel = lane>>4;
//             element e -> K = (e&7) + ((e>>3)<<4) + 8*halfsel  (interleaved)
//  B (32x16): lane -> col N = lane&15, halfsel = lane>>4;
//             element e -> K = (e&7) + ((e>>3)<<4) + 8*halfsel                   (as A)
//  C/D (16x16 f32): VGPR r -> M = r + 8*halfsel; lane -> N = lane&15
// All LDS tiles are stored PRE-SWIZZLED so a lane's 16 fragment halves are
// 32 contiguous bytes -> fragment loads are 2x ds_load_b128.
// ---------------------------------------------------------------------------

// ---------------------------------------------------------------------------
// Tiled WMMA GEMM: C[z] = A[z/hdiv] (MxK) @ B[z%hdiv] (KxN)
// Block = 128 threads (4 waves); block tile 64(M) x 64(N); K step 32.
// Wave w owns M sub-tile w and iterates the 4 N sub-tiles: 4 WMMA per K-step.
// Grid: (N/64, M/64, Z)
// ---------------------------------------------------------------------------
template<int A_F32, int C_F16, int C_TRANS>
__global__ __launch_bounds__(128)
void wmma_gemm_kernel(const void* __restrict__ Ap, long strideA,
                      const float* __restrict__ Bp, long strideB,
                      void* __restrict__ Cp, long strideC,
                      int K, int lda, int ldb, int ldc, int hdiv)
{
    // Swizzled tiles: [subtile][lane][16 halves]
    __shared__ __attribute__((aligned(32))) _Float16 As_sw[4][32][16];  // 4 KB
    __shared__ __attribute__((aligned(32))) _Float16 Bs_sw[4][32][16];  // 4 KB

    const int tid    = threadIdx.x;
    const int lane   = tid & 31;
    const int wid    = tid >> 5;            // wave id -> M sub-tile
    const int row    = lane & 15;
    const int half16 = lane >> 4;

    const int  z    = blockIdx.z;
    const int  m0   = blockIdx.y * 64;
    const int  n0   = blockIdx.x * 64;
    const long aOff = (long)(z / hdiv) * strideA;
    const long bOff = (long)(z % hdiv) * strideB;
    const long cOff = (long)z * strideC;

    // A staging: thread -> 16 consecutive k of one row (2 threads per row).
    //  k chunk [akh*16, akh*16+16): lower 8 -> halfsel 0, upper 8 -> halfsel 1,
    //  both at element offset akh*8 (per interleave map).
    const int arr = tid >> 1;                // 0..63 row in 64-row tile
    const int akh = tid & 1;                 // which 16-k chunk
    const int amw = arr >> 4;                // owning M sub-tile
    const int ar  = arr & 15;
    // B staging: thread -> 16 k (8*bkh+{0..7,16..23}) at fixed col (one 32B LDS store)
    const int bc  = tid & 63;                // col
    const int bkh = tid >> 6;                // k half (0..1)

    v8f acc[4] = {};

    for (int k0 = 0; k0 < K; k0 += 32) {
        // ---- stage A (f32/f16 -> f16, swizzled) ----
        {
            long gi = aOff + (long)(m0 + arr) * lda + (k0 + akh * 16);
            v8h loh, hih;
            if (A_F32) {
                const float* af = &((const float*)Ap)[gi];
                v4f f0 = *(const v4f*)(af + 0);
                v4f f1 = *(const v4f*)(af + 4);
                v4f f2 = *(const v4f*)(af + 8);
                v4f f3 = *(const v4f*)(af + 12);
                #pragma unroll
                for (int i = 0; i < 4; ++i) {
                    loh[i]     = (_Float16)f0[i];
                    loh[i + 4] = (_Float16)f1[i];
                    hih[i]     = (_Float16)f2[i];
                    hih[i + 4] = (_Float16)f3[i];
                }
            } else {
                const _Float16* ah = &((const _Float16*)Ap)[gi];
                loh = *(const v8h*)(ah);
                hih = *(const v8h*)(ah + 8);
            }
            *(v8h*)&As_sw[amw][ar][akh * 8]      = loh;  // halfsel 0 lanes
            *(v8h*)&As_sw[amw][16 + ar][akh * 8] = hih;  // halfsel 1 lanes
        }
        // ---- stage B (f32 -> f16, swizzled) ----
        {
            v16h bt;
            #pragma unroll
            for (int i = 0; i < 16; ++i) {
                long gi = bOff + (long)(k0 + bkh * 8 + (i & 7) + ((i >> 3) << 4)) * ldb + (n0 + bc);
                bt[i] = (_Float16)Bp[gi];
            }
            *(v16h*)&Bs_sw[bc >> 4][bkh * 16 + (bc & 15)][0] = bt;
        }
        __syncthreads();

        v16h a = *(const v16h*)&As_sw[wid][lane][0];
        #pragma unroll
        for (int ns = 0; ns < 4; ++ns) {
            v16h b = *(const v16h*)&Bs_sw[ns][lane][0];
            acc[ns] = __builtin_amdgcn_wmma_f32_16x16x32_f16(
                          false, a, false, b, (short)0, acc[ns], false, false);
        }
        __syncthreads();
    }

    #pragma unroll
    for (int ns = 0; ns < 4; ++ns)
        #pragma unroll
        for (int r = 0; r < 8; ++r) {
            int  mi = wid * 16 + r + 8 * half16;
            int  ni = ns * 16 + row;
            long ci = C_TRANS ? (cOff + (long)(n0 + ni) * ldc + (m0 + mi))
                              : (cOff + (long)(m0 + mi) * ldc + (n0 + ni));
            if (C_F16) ((_Float16*)Cp)[ci] = (_Float16)acc[ns][r];
            else       ((float*)Cp)[ci]    = acc[ns][r];
        }
}

// ---------------------------------------------------------------------------
// Fused attention. softmax(axis=0) couples only the 8 batch values at each
// (h,i,j) -> QK, softmax, PV fuse with no logits in HBM.
// Block = 256 threads = 8 waves; wave w owns batch b = w.
// Grid: (H, I/16); loop j in steps of 32.
// ---------------------------------------------------------------------------
#define NBATCH 8
#define NHEAD  16
#define SEQ    1024
#define HD     64

__global__ __launch_bounds__(256)
void attn_fused_kernel(const _Float16* __restrict__ wq,
                       const _Float16* __restrict__ wk,
                       const _Float16* __restrict__ wvT,   // [b][h][s][j]
                       _Float16* __restrict__ heads)
{
    // 32 KB buffer time-shared: K-phase tiles then V-phase tiles (swizzled)
    __shared__ __attribute__((aligned(32))) _Float16 kv[NBATCH * 2048];
    // logit exchange: [t][r][lane][b] -> per-lane v8f read of all 8 batches
    __shared__ __attribute__((aligned(32))) float smx[2][8][32][NBATCH]; // 16 KB
    // probs staged in A-fragment layout: [b][lane][e]
    __shared__ __attribute__((aligned(32))) _Float16 paS[NBATCH][32][16]; // 8 KB

    const int h      = blockIdx.x;
    const int i0     = blockIdx.y * 16;
    const int lane   = threadIdx.x & 31;
    const int b      = threadIdx.x >> 5;    // wave id == batch id
    const int row    = lane & 15;
    const int half16 = lane >> 4;

    const long bh    = (long)b * NHEAD + h;
    const long kbase = bh * SEQ;            // row-block base for (b,h)

    // Preload wq A-fragments: 16x64 = two 16x32 frags, contiguous 16B chunks
    const _Float16* qrow = wq + (kbase + i0 + row) * HD;
    v8h lo0 = *(const v8h*)(qrow + 16 * half16);
    v8h hi0 = *(const v8h*)(qrow + 8 + 16 * half16);
    v8h lo1 = *(const v8h*)(qrow + 32 + 16 * half16);
    v8h hi1 = *(const v8h*)(qrow + 40 + 16 * half16);
    v16h aq0 = __builtin_shufflevector(lo0, hi0, 0,1,2,3,4,5,6,7,8,9,10,11,12,13,14,15);
    v16h aq1 = __builtin_shufflevector(lo1, hi1, 0,1,2,3,4,5,6,7,8,9,10,11,12,13,14,15);

    v8f accs[4] = {};   // head tile 16(i) x 64(s)

    for (int j0 = 0; j0 < SEQ; j0 += 32) {
        // ---- stage wk tiles (wave b stages its own batch; lane = j row) ----
        {
            const _Float16* src = wk + (kbase + j0 + lane) * HD;
            v16h c0 = *(const v16h*)(src);        // s  0..15
            v16h c1 = *(const v16h*)(src + 16);   // s 16..31
            v16h c2 = *(const v16h*)(src + 32);   // s 32..47
            v16h c3 = *(const v16h*)(src + 48);   // s 48..63
            int tt = lane >> 4, nn = lane & 15;
            _Float16* d = &kv[((b * 2 + tt) * 32) * 32];
            *(v16h*)&d[(nn)      * 32 + 0]  = c0;
            *(v16h*)&d[(16 + nn) * 32 + 0]  = c1;
            *(v16h*)&d[(nn)      * 32 + 16] = c2;
            *(v16h*)&d[(16 + nn) * 32 + 16] = c3;
        }
        __syncthreads();

        // ---- QK logits: two 16x16 j sub-tiles, K = 64 over s ----
        v8f L[2];
        #pragma unroll
        for (int t = 0; t < 2; ++t) {
            const _Float16* kb_ = &kv[((b * 2 + t) * 32 + lane) * 32];
            v16h bk0 = *(const v16h*)(kb_);
            v16h bk1 = *(const v16h*)(kb_ + 16);
            v8f l = {};
            l = __builtin_amdgcn_wmma_f32_16x16x32_f16(
                    false, aq0, false, bk0, (short)0, l, false, false);
            l = __builtin_amdgcn_wmma_f32_16x16x32_f16(
                    false, aq1, false, bk1, (short)0, l, false, false);
            L[t] = l;
            #pragma unroll
            for (int r = 0; r < 8; ++r)
                smx[t][r][lane][b] = l[r] * 0.25f;     // quirk: /sqrt(H)=4
        }
        __syncthreads();

        // ---- softmax over the 8 batches; scatter probs into A-frag layout ----
        #pragma unroll
        for (int t = 0; t < 2; ++t) {
            #pragma unroll
            for (int r = 0; r < 8; ++r) {
                v8f vals = *(const v8f*)&smx[t][r][lane][0];  // 2x ds_load_b128
                float m = vals[0];
                #pragma unroll
                for (int bb = 1; bb < NBATCH; ++bb) m = fmaxf(m, vals[bb]);
                float s = 0.f;
                #pragma unroll
                for (int bb = 0; bb < NBATCH; ++bb) s += __expf(vals[bb] - m);
                float w = __expf(L[t][r] * 0.25f - m) / s;
                int mi = r + 8 * half16;
                int jl = t * 16 + row;
                int dl = mi + 16 * (jl >> 4);
                int de = jl & 15;
                paS[b][dl][de] = (_Float16)w;
            }
        }
        __syncthreads();   // smx reads done; kv K-phase reads done -> reuse kv

        // ---- stage wvT tiles (lane -> 2 s rows; contiguous j chunks) ----
        {
            #pragma unroll
            for (int q = 0; q < 2; ++q) {
                int s = lane * 2 + q;
                const _Float16* vsrc = wvT + (bh * HD + s) * SEQ + j0;
                v16h d0 = *(const v16h*)(vsrc);        // j  0..15
                v16h d1 = *(const v16h*)(vsrc + 16);   // j 16..31
                int st = s >> 4, nn = s & 15;
                _Float16* d = &kv[((b * 4 + st) * 32) * 16];
                *(v16h*)&d[(nn)      * 16] = d0;
                *(v16h*)&d[(16 + nn) * 16] = d1;
            }
        }
        __syncthreads();

        // ---- PV: probs (16x32) @ wv_tile (32x64), accumulate ----
        v16h pa = *(const v16h*)&paS[b][lane][0];
        #pragma unroll
        for (int st = 0; st < 4; ++st) {
            v16h bv = *(const v16h*)&kv[((b * 4 + st) * 32 + lane) * 16];
            accs[st] = __builtin_amdgcn_wmma_f32_16x16x32_f16(
                           false, pa, false, bv, (short)0, accs[st], false, false);
        }
        __syncthreads();   // kv V-phase reads done before next K staging
    }

    // heads stored contiguously as [b][h][i][s] f16; the torch .view quirk then
    // makes the output projection a plain [b][1024][1024] row-major GEMM.
    #pragma unroll
    for (int st = 0; st < 4; ++st)
        #pragma unroll
        for (int r = 0; r < 8; ++r) {
            int mi = r + 8 * half16;
            int ni = st * 16 + row;
            heads[(kbase + i0 + mi) * HD + ni] = (_Float16)accs[st][r];
        }
}

// ---------------------------------------------------------------------------
extern "C" void kernel_launch(void* const* d_in, const int* in_sizes, int n_in,
                              void* d_out, int out_size, void* d_ws, size_t ws_size,
                              hipStream_t stream) {
    const float* query = (const float*)d_in[0];   // [8,1024,1024]
    const float* key   = (const float*)d_in[1];   // [8,1024,1024]
    const float* value = (const float*)d_in[2];   // [8,1024,1024]
    const float* q_w   = (const float*)d_in[3];   // [16,1024,64]
    const float* k_w   = (const float*)d_in[4];   // [16,1024,64]
    const float* v_w   = (const float*)d_in[5];   // [16,1024,64]
    const float* o_w   = (const float*)d_in[6];   // [1024,1024]
    float* out = (float*)d_out;                   // [8,1024,1024]

    const size_t projElems = (size_t)NBATCH * NHEAD * SEQ * HD;  // 8.39M halves
    _Float16* wq    = (_Float16*)d_ws;
    _Float16* wk    = wq + projElems;
    _Float16* wvT   = wk + projElems;             // transposed: [b][h][s][j]
    _Float16* heads = wvT + projElems;            // total ws: 64 MB f16

    const dim3 blk(128);
    const long sA = (long)SEQ * 1024;     // per-batch input stride
    const long sB = (long)1024 * HD;      // per-head weight stride
    const long sC = (long)SEQ * HD;       // per-(b,h) proj stride

    // Projections: C[z=b*16+h] = X[b](1024x1024) @ W[h](1024x64), f16 out
    wmma_gemm_kernel<1, 1, 0><<<dim3(1, SEQ / 64, NBATCH * NHEAD), blk, 0, stream>>>(
        query, sA, q_w, sB, wq, sC, 1024, 1024, HD, HD, NHEAD);
    wmma_gemm_kernel<1, 1, 0><<<dim3(1, SEQ / 64, NBATCH * NHEAD), blk, 0, stream>>>(
        key, sA, k_w, sB, wk, sC, 1024, 1024, HD, HD, NHEAD);
    // V projection stored transposed ([s][j]) so attention V staging is
    // contiguous (PV B-fragments want contiguous j at fixed s).
    wmma_gemm_kernel<1, 1, 1><<<dim3(1, SEQ / 64, NBATCH * NHEAD), blk, 0, stream>>>(
        value, sA, v_w, sB, wvT, sC, 1024, 1024, HD, SEQ, NHEAD);

    // Fused QK -> batch-axis softmax -> PV
    attn_fused_kernel<<<dim3(NHEAD, SEQ / 16), 256, 0, stream>>>(wq, wk, wvT, heads);

    // Output projection: out[b] = headsFlat[b](1024x1024 f16) @ o_w(1024x1024)
    wmma_gemm_kernel<0, 0, 0><<<dim3(1024 / 64, 1024 / 64, NBATCH), blk, 0, stream>>>(
        heads, (long)SEQ * 1024, o_w, 0L,
        out, (long)SEQ * 1024, 1024, 1024, 1024, 1024, 1);
}
